// CausalSelfAttention_36155034698430
// MI455X (gfx1250) — matrix operand order repaired and checked by reading, not compiled, not run on hardware
//
#include <hip/hip_runtime.h>
#include <hip/hip_bf16.h>

typedef __attribute__((ext_vector_type(16))) __bf16 bf16x16;
typedef __attribute__((ext_vector_type(8)))  __bf16 bf16x8;
typedef __attribute__((ext_vector_type(8)))  float  f32x8;
typedef __attribute__((ext_vector_type(4)))  float  f32x4;

static constexpr int BATCH = 2;
static constexpr int T = 4096;
static constexpr int C = 768;
static constexpr int H = 12;
static constexpr int D = 64;

#define DEVI static __device__ __forceinline__

DEVI __bf16 f2bf(float f) {
    unsigned u = __builtin_bit_cast(unsigned, f);
    u += 0x7FFFu + ((u >> 16) & 1u);          // round-to-nearest-even
    unsigned short h = (unsigned short)(u >> 16);
    return __builtin_bit_cast(__bf16, h);
}

DEVI f32x8 wmma_bf16(bf16x16 a, bf16x16 b, f32x8 c) {
    return __builtin_amdgcn_wmma_f32_16x16x32_bf16(
        false, a, false, b, (short)0, c, false, false);
}

// Build a v16 bf16 WMMA operand from two 16-byte segments (global or LDS).
DEVI bf16x16 ld16bf(const __bf16* p0, const __bf16* p1) {
    bf16x8 a = *(const bf16x8*)p0;
    bf16x8 b = *(const bf16x8*)p1;
    bf16x16 r;
#pragma unroll
    for (int i = 0; i < 8; ++i) { r[i] = a[i]; r[i + 8] = b[i]; }
    return r;
}

// ---------------------------------------------------------------------------
// Kernel 0: one-shot f32 -> bf16 conversion (x and the four weight matrices).
// Converts once so the GEMM inner loops are pure b128 loads + WMMA.
// ---------------------------------------------------------------------------
__global__ __launch_bounds__(256) void cvt_bf16(
    const float* __restrict__ src, __bf16* __restrict__ dst, int n8)
{
    const int i = blockIdx.x * 256 + threadIdx.x;
    if (i >= n8) return;
    const f32x4 a = ((const f32x4*)src)[2 * i];
    const f32x4 b = ((const f32x4*)src)[2 * i + 1];
    bf16x8 o;
#pragma unroll
    for (int j = 0; j < 4; ++j) { o[j] = f2bf(a[j]); o[4 + j] = f2bf(b[j]); }
    ((bf16x8*)dst)[i] = o;
}

// ---------------------------------------------------------------------------
// Kernel 1: Q/K/V projections.  Q,K stored bf16 [B,H,T,D]; V stored bf16
// transposed [B,H,D,T] so att@V B-operands are contiguous b128 loads.
// One wave per 16x16 output tile; K-dim = C = 768 = 24 WMMA steps.
// ---------------------------------------------------------------------------
__global__ __launch_bounds__(256) void qkv_wmma(
    const __bf16* __restrict__ x,
    const __bf16* __restrict__ Wq, const __bf16* __restrict__ Wk,
    const __bf16* __restrict__ Wv,
    __bf16* __restrict__ Qb, __bf16* __restrict__ Kb, __bf16* __restrict__ Vt)
{
    const int lane = threadIdx.x & 31;
    const int hlf  = lane >> 4;           // which 16-lane half
    const int ln   = lane & 15;
    const int wid  = threadIdx.x >> 5;

    const int tile = blockIdx.x * 8 + wid;        // 512*48 tiles total
    const int tb   = tile / (C / 16);
    const int jb   = tile % (C / 16);
    const int row0 = tb * 16;                     // global row in [0, B*T)

    const int arow = row0 + ln;                   // A-operand row (M = lane%16)
    const __bf16* xr  = x  + (size_t)arow * C;
    const int j       = jb * 16 + ln;             // output column (N = lane%16)
    const __bf16* wqr = Wq + (size_t)j * C;
    const __bf16* wkr = Wk + (size_t)j * C;
    const __bf16* wvr = Wv + (size_t)j * C;

    f32x8 aq, ak, av;
#pragma unroll
    for (int i = 0; i < 8; ++i) { aq[i] = 0.f; ak[i] = 0.f; av[i] = 0.f; }

    for (int cc = 0; cc < C; cc += 32) {
        // A (x): half0 holds K {0..7,16..23}, half1 holds K {8..15,24..31}
        bf16x16 A  = ld16bf(xr + cc + hlf * 8, xr + cc + 16 + hlf * 8);
        // B (W^T): same K split as A: half0 K {0..7,16..23}, half1 K {8..15,24..31}
        bf16x16 Bq = ld16bf(wqr + cc + hlf * 8, wqr + cc + 16 + hlf * 8);
        bf16x16 Bk = ld16bf(wkr + cc + hlf * 8, wkr + cc + 16 + hlf * 8);
        bf16x16 Bv = ld16bf(wvr + cc + hlf * 8, wvr + cc + 16 + hlf * 8);
        aq = wmma_bf16(A, Bq, aq);
        ak = wmma_bf16(A, Bk, ak);
        av = wmma_bf16(A, Bv, av);
    }

    const int hh = j / D, d = j % D;
#pragma unroll
    for (int r = 0; r < 8; ++r) {                 // C layout: M = r + 8*half
        const int row = row0 + r + 8 * hlf;
        const int b = row / T, t = row % T;
        const size_t qidx = (((size_t)(b * H + hh) * T + t) * D + d);
        Qb[qidx] = f2bf(aq[r]);
        Kb[qidx] = f2bf(ak[r]);
        Vt[(((size_t)(b * H + hh) * D + d) * T + t)] = f2bf(av[r]);
    }
}

// ---------------------------------------------------------------------------
// Kernel 2: per (b,h,16-row q tile): scores WMMA + causal mask + 2-pass
// softmax (recompute), NT-stream att, att@V WMMA, normalize + zero-fill.
// ---------------------------------------------------------------------------
__global__ __launch_bounds__(256) void attn_wmma(
    const __bf16* __restrict__ Q, const __bf16* __restrict__ K,
    const __bf16* __restrict__ Vt,
    float* __restrict__ att, __bf16* __restrict__ Yh)
{
    __shared__ __attribute__((aligned(16))) unsigned short elds[8][16 * 40];

    const int lane = threadIdx.x & 31;
    const int hlf  = lane >> 4;
    const int ln   = lane & 15;
    const int wid  = threadIdx.x >> 5;

    const int u  = blockIdx.x * 8 + wid;          // 2*12*256 units
    const int qt = u & 255;
    const int hh = (u >> 8) % H;
    const int b  = u / (256 * H);
    const int qbase = qt * 16;

    const __bf16* Qh = Q  + (size_t)(b * H + hh) * T * D;
    const __bf16* Kh = K  + (size_t)(b * H + hh) * T * D;
    const __bf16* Vh = Vt + (size_t)(b * H + hh) * D * T;
    float*  attR = att + (size_t)(b * H + hh) * T * T;
    __bf16* Yhh  = Yh  + (size_t)(b * H + hh) * T * D;

    // Q tile A-operands for both 32-wide d-chunks (D = 64)
    bf16x16 Aq[2];
#pragma unroll
    for (int ch = 0; ch < 2; ++ch) {
        const __bf16* p = Qh + (size_t)(qbase + ln) * D + ch * 32;
        Aq[ch] = ld16bf(p + hlf * 8, p + 16 + hlf * 8);
    }

    const int nk = ((qbase + 47) >> 5) << 5;      // keys processed (mult of 32)

    // ---- pass 1: row max -------------------------------------------------
    float m[8];
#pragma unroll
    for (int r = 0; r < 8; ++r) m[r] = -3.0e38f;

    for (int kb = 0; kb < nk; kb += 16) {
        f32x8 s;
#pragma unroll
        for (int i = 0; i < 8; ++i) s[i] = 0.f;
#pragma unroll
        for (int ch = 0; ch < 2; ++ch) {
            const __bf16* p = Kh + (size_t)(kb + ln) * D + ch * 32;
            s = wmma_bf16(Aq[ch], ld16bf(p + hlf * 8, p + 16 + hlf * 8), s);
        }
#pragma unroll
        for (int r = 0; r < 8; ++r) {
            const int q = qbase + r + 8 * hlf, k = kb + ln;
            m[r] = fmaxf(m[r], (k <= q) ? s[r] * 0.125f : -3.0e38f);
        }
    }
#pragma unroll
    for (int r = 0; r < 8; ++r)
#pragma unroll
        for (int msk = 1; msk < 16; msk <<= 1)
            m[r] = fmaxf(m[r], __shfl_xor(m[r], msk, 32));

    // ---- pass 2: exp, NT-store att, accumulate att@V ---------------------
    float rs[8];
#pragma unroll
    for (int r = 0; r < 8; ++r) rs[r] = 0.f;
    f32x8 yacc[4];
#pragma unroll
    for (int nt = 0; nt < 4; ++nt)
#pragma unroll
        for (int i = 0; i < 8; ++i) yacc[nt][i] = 0.f;

    for (int kb = 0; kb < nk; kb += 32) {
        if (kb + 32 < nk)
            __builtin_prefetch(Kh + (size_t)(kb + 32 + ln) * D, 0, 1);
#pragma unroll
        for (int t2 = 0; t2 < 2; ++t2) {
            const int kb16 = kb + t2 * 16;
            f32x8 s;
#pragma unroll
            for (int i = 0; i < 8; ++i) s[i] = 0.f;
#pragma unroll
            for (int ch = 0; ch < 2; ++ch) {
                const __bf16* p = Kh + (size_t)(kb16 + ln) * D + ch * 32;
                s = wmma_bf16(Aq[ch], ld16bf(p + hlf * 8, p + 16 + hlf * 8), s);
            }
#pragma unroll
            for (int r = 0; r < 8; ++r) {
                const int q = qbase + r + 8 * hlf, k = kb16 + ln;
                const float e = (k <= q) ? __expf(s[r] * 0.125f - m[r]) : 0.0f;
                rs[r] += e;
                __builtin_nontemporal_store(e, attR + (size_t)q * T + k);
                elds[wid][(r + 8 * hlf) * 40 + t2 * 16 + ln] =
                    __builtin_bit_cast(unsigned short, f2bf(e));
            }
        }
        // C-layout -> A-layout reshuffle through per-wave LDS (same-wave DS order)
        const unsigned short* lb = &elds[wid][ln * 40];
        bf16x16 Ae = ld16bf((const __bf16*)(lb + hlf * 8),
                            (const __bf16*)(lb + 16 + hlf * 8));
#pragma unroll
        for (int nt = 0; nt < 4; ++nt) {
            const __bf16* p = Vh + (size_t)(nt * 16 + ln) * T + kb;
            yacc[nt] = wmma_bf16(Ae, ld16bf(p + hlf * 8, p + 16 + hlf * 8), yacc[nt]);
        }
    }

    // ---- row sums, scale y, store bf16 head output -----------------------
#pragma unroll
    for (int r = 0; r < 8; ++r)
#pragma unroll
        for (int msk = 1; msk < 16; msk <<= 1)
            rs[r] += __shfl_xor(rs[r], msk, 32);
    float inv[8];
#pragma unroll
    for (int r = 0; r < 8; ++r) inv[r] = 1.0f / rs[r];

#pragma unroll
    for (int nt = 0; nt < 4; ++nt)
#pragma unroll
        for (int r = 0; r < 8; ++r)
            Yhh[(size_t)(qbase + r + 8 * hlf) * D + nt * 16 + ln] =
                f2bf(yacc[nt][r] * inv[r]);

    // ---- pass 3: normalize written att rows, zero-fill causal tail -------
    float invRow[16];
#pragma unroll
    for (int q2 = 0; q2 < 16; ++q2)
        invRow[q2] = __shfl(inv[q2 & 7], (q2 >> 3) << 4, 32);

    f32x4 z;
#pragma unroll
    for (int i = 0; i < 4; ++i) z[i] = 0.f;

    for (int q2 = 0; q2 < 16; ++q2) {
        float* rp = attR + (size_t)(qbase + q2) * T;
        const float sc = invRow[q2];
        for (int k = lane * 4; k < nk; k += 128) {
            f32x4 v = *(const f32x4*)(rp + k);
            v = v * sc;
            __builtin_nontemporal_store(v, (f32x4*)(rp + k));
        }
        for (int k = nk + lane * 4; k < T; k += 128)
            __builtin_nontemporal_store(z, (f32x4*)(rp + k));
    }
}

// ---------------------------------------------------------------------------
// Kernel 3: y = merge(Yh) @ Wp^T + bp   (Yh is bf16 [B,H,T,D])
// ---------------------------------------------------------------------------
__global__ __launch_bounds__(256) void proj_wmma(
    const __bf16* __restrict__ Yh, const __bf16* __restrict__ Wp,
    const float* __restrict__ bp, float* __restrict__ y)
{
    const int lane = threadIdx.x & 31;
    const int hlf  = lane >> 4;
    const int ln   = lane & 15;
    const int wid  = threadIdx.x >> 5;

    const int tile = blockIdx.x * 8 + wid;
    const int tb   = tile / (C / 16);
    const int jb   = tile % (C / 16);
    const int row0 = tb * 16;

    const int arow = row0 + ln;
    const int b = arow / T, t = arow % T;
    const int j = jb * 16 + ln;
    const __bf16* wpr = Wp + (size_t)j * C;

    f32x8 acc;
#pragma unroll
    for (int i = 0; i < 8; ++i) acc[i] = 0.f;

    for (int cc = 0; cc < C; cc += 32) {
        const int c0 = cc + hlf * 8, c1 = cc + 16 + hlf * 8;
        // merged view: col c -> head c/64, dim c%64 (8-elem segs never cross heads)
        const __bf16* p0 = Yh + (((size_t)(b * H + c0 / D) * T + t) * D + c0 % D);
        const __bf16* p1 = Yh + (((size_t)(b * H + c1 / D) * T + t) * D + c1 % D);
        bf16x16 A  = ld16bf(p0, p1);
        bf16x16 Bp = ld16bf(wpr + cc + hlf * 8, wpr + cc + 16 + hlf * 8);
        acc = wmma_bf16(A, Bp, acc);
    }

    const float bias = bp[j];
#pragma unroll
    for (int r = 0; r < 8; ++r) {
        const int row = row0 + r + 8 * hlf;
        y[(size_t)row * C + j] = acc[r] + bias;
    }
}

// ---------------------------------------------------------------------------
extern "C" void kernel_launch(void* const* d_in, const int* in_sizes, int n_in,
                              void* d_out, int out_size, void* d_ws, size_t ws_size,
                              hipStream_t stream) {
    (void)in_sizes; (void)n_in; (void)out_size; (void)ws_size;
    const float* x  = (const float*)d_in[0];
    const float* Wq = (const float*)d_in[1];
    const float* Wk = (const float*)d_in[2];
    const float* Wv = (const float*)d_in[3];
    const float* Wp = (const float*)d_in[4];
    const float* bp = (const float*)d_in[5];

    float* y   = (float*)d_out;
    float* att = y + (size_t)BATCH * T * C;       // outputs concatenated flat

    const size_t xN   = (size_t)BATCH * T * C;    // 6,291,456
    const size_t wN   = (size_t)C * C;            //   589,824
    const size_t qkvN = (size_t)BATCH * H * T * D;// 6,291,456

    __bf16* xb  = (__bf16*)d_ws;                  // 12.6 MB
    __bf16* Wqb = xb  + xN;                       //  1.2 MB
    __bf16* Wkb = Wqb + wN;
    __bf16* Wvb = Wkb + wN;
    __bf16* Wpb = Wvb + wN;
    __bf16* Qb  = Wpb + wN;                       // 12.6 MB
    __bf16* Kb  = Qb  + qkvN;                     // 12.6 MB
    __bf16* Vt  = Kb  + qkvN;                     // 12.6 MB (transposed V)
    __bf16* Yh  = Vt  + qkvN;                     // 12.6 MB (bf16 head output)

    // one-shot f32 -> bf16 conversions (34 MB total, trivial vs 1.6 GB att)
    cvt_bf16<<<(int)(xN / 8 + 255) / 256, 256, 0, stream>>>(x,  xb,  (int)(xN / 8));
    cvt_bf16<<<(int)(wN / 8 + 255) / 256, 256, 0, stream>>>(Wq, Wqb, (int)(wN / 8));
    cvt_bf16<<<(int)(wN / 8 + 255) / 256, 256, 0, stream>>>(Wk, Wkb, (int)(wN / 8));
    cvt_bf16<<<(int)(wN / 8 + 255) / 256, 256, 0, stream>>>(Wv, Wvb, (int)(wN / 8));
    cvt_bf16<<<(int)(wN / 8 + 255) / 256, 256, 0, stream>>>(Wp, Wpb, (int)(wN / 8));

    const int gemmTiles = (BATCH * T / 16) * (C / 16);  // 24576 -> 3072 blocks
    const int attnUnits = BATCH * H * (T / 16);         //  6144 ->  768 blocks

    qkv_wmma <<<gemmTiles / 8, 256, 0, stream>>>(xb, Wqb, Wkb, Wvb, Qb, Kb, Vt);
    attn_wmma<<<attnUnits / 8, 256, 0, stream>>>(Qb, Kb, Vt, att, Yh);
    proj_wmma<<<gemmTiles / 8, 256, 0, stream>>>(Yh, Wpb, bp, y);
}
